// HybridQLSTMKernel_65481071398570
// MI455X (gfx1250) — matrix operand order repaired and checked by reading, not compiled, not run on hardware
//
#include <hip/hip_runtime.h>
#include <hip/hip_bf16.h>

typedef __attribute__((ext_vector_type(16))) __bf16        v16bf;
typedef __attribute__((ext_vector_type(8)))  float         v8f;
typedef __attribute__((ext_vector_type(8)))  unsigned int  v8u;

#define SEQ    2048
#define EMB    1024
#define NQ     4
#define TAGS   64
#define KTILES (EMB / 32)   // 32 k-tiles of 32
#define MTILES (SEQ / 16)   // 128 row-tiles of 16

// ---------------------------------------------------------------------------
// K0: pack B = [W_f | W_i | W_u | W_o] (1024 x 16, f32, row stride NQ) into
// bf16 WMMA B-fragments for all 32 K-tiles.
// B-register layout (16x16x32 bf16): lane 0-15 -> N=lane, K=ktile*32 + 0..7,16..23
//                                    lane 16-31 -> N=lane-16, K=ktile*32+8..15,24..31
// packed 2 bf16 per dword, 8 dwords (32B) per lane per tile.
// ---------------------------------------------------------------------------
__global__ void __launch_bounds__(32) qlstm_pack_b(
    const float* __restrict__ Wf, const float* __restrict__ Wi,
    const float* __restrict__ Wu, const float* __restrict__ Wo,
    unsigned int* __restrict__ bfrag /* [KTILES][32][8] dwords */) {
  int kt   = blockIdx.x;
  int lane = threadIdx.x;
  int n = lane & 15;
  int q = n & 3;
  const float* Wg = (n < 4) ? Wf : (n < 8) ? Wi : (n < 12) ? Wu : Wo;
  int k0 = kt * 32 + ((lane >= 16) ? 8 : 0);
  unsigned int* dst = bfrag + (kt * 32 + lane) * 8;
#pragma unroll
  for (int d = 0; d < 8; ++d) {
    __bf16 lo = (__bf16)Wg[(k0 + 2 * d + (d >= 4 ? 8 : 0)    ) * NQ + q];
    __bf16 hi = (__bf16)Wg[(k0 + 2 * d + (d >= 4 ? 8 : 0) + 1) * NQ + q];
    unsigned int w = (unsigned int)__builtin_bit_cast(unsigned short, lo)
                   | ((unsigned int)__builtin_bit_cast(unsigned short, hi) << 16);
    dst[d] = w;
  }
}

// ---------------------------------------------------------------------------
// K1: fused embedding-gather + GEMM via v_wmma_f32_16x16x32_bf16.
// One wave per 16 timesteps. A-fragment layout (ISA 7.12.2, 16-bit A 16x32):
//   lane 0-15 : M=lane,    halfwords 0-7 -> K=0..7,  8-15 -> K=16..23
//   lane 16-31: M=lane-16, halfwords 0-7 -> K=8..15, 8-15 -> K=24..31
// Output pre[t][n] = x_t @ Wcol_n + b + phi  (bias & RY angle folded in).
// ---------------------------------------------------------------------------
__global__ void __launch_bounds__(32) qlstm_gemm_pre(
    const int* __restrict__ sentence, const float* __restrict__ emb,
    const unsigned int* __restrict__ bfrag,
    const float* __restrict__ bf_, const float* __restrict__ bi_,
    const float* __restrict__ bu_, const float* __restrict__ bo_,
    const float* __restrict__ pf_, const float* __restrict__ pi_,
    const float* __restrict__ pu_, const float* __restrict__ po_,
    float* __restrict__ pre /* [SEQ][16] */) {
  int lane = threadIdx.x;
  bool hi  = lane >= 16;
  int  mr  = lane & 15;               // A-row in tile == B-column n
  int  t0  = blockIdx.x * 16;

  int token = sentence[t0 + mr];
  const float* arow = emb + (size_t)token * EMB;

  int q = mr & 3;
  const float* bg = (mr < 4) ? bf_ : (mr < 8) ? bi_ : (mr < 12) ? bu_ : bo_;
  const float* pg = (mr < 4) ? pf_ : (mr < 8) ? pi_ : (mr < 12) ? pu_ : po_;
  float colbias = bg[q] + pg[q];      // fold b and phi: cos(xW + b + hWh + phi)

  const v8u* bp = (const v8u*)bfrag;
  int aoff = hi ? 8 : 0;

  v8f acc = {};
  for (int kt = 0; kt < KTILES; ++kt) {
    const float* ap = arow + kt * 32 + aoff;          // 16B-aligned
    float4 a0 = *(const float4*)(ap);
    float4 a1 = *(const float4*)(ap + 4);
    float4 a2 = *(const float4*)(ap + 16);
    float4 a3 = *(const float4*)(ap + 20);
    v16bf av;
    av[0]  = (__bf16)a0.x; av[1]  = (__bf16)a0.y; av[2]  = (__bf16)a0.z; av[3]  = (__bf16)a0.w;
    av[4]  = (__bf16)a1.x; av[5]  = (__bf16)a1.y; av[6]  = (__bf16)a1.z; av[7]  = (__bf16)a1.w;
    av[8]  = (__bf16)a2.x; av[9]  = (__bf16)a2.y; av[10] = (__bf16)a2.z; av[11] = (__bf16)a2.w;
    av[12] = (__bf16)a3.x; av[13] = (__bf16)a3.y; av[14] = (__bf16)a3.z; av[15] = (__bf16)a3.w;

    v8u   braw = bp[kt * 32 + lane];                  // 32B coalesced
    v16bf bv   = __builtin_bit_cast(v16bf, braw);

    acc = __builtin_amdgcn_wmma_f32_16x16x32_bf16(
        /*neg_a=*/false, av, /*neg_b=*/false, bv,
        /*c_mod=*/(short)0, acc, /*reuse_a=*/false, /*reuse_b=*/false);
  }

  // C/D layout: VGPR r -> M=r (lanes 0-15), M=r+8 (lanes 16-31); N=lane&15
  int mbase = t0 + (hi ? 8 : 0);
#pragma unroll
  for (int r = 0; r < 8; ++r)
    pre[(size_t)(mbase + r) * 16 + mr] = acc[r] + colbias;
}

// ---------------------------------------------------------------------------
// K2: sequential LSTM scan, one wave. Lane n (0..15) owns gate-column n
// (gate g=n/4, unit q=n%4). Cross-lane traffic via wave32 shuffles.
// ---------------------------------------------------------------------------
__global__ void __launch_bounds__(32) qlstm_scan(
    const float* __restrict__ pre,
    const float* __restrict__ Wf, const float* __restrict__ Wi,
    const float* __restrict__ Wu, const float* __restrict__ Wo,
    float* __restrict__ hout /* [SEQ][4] */) {
  int lane = threadIdx.x;
  int n = lane & 15;
  int q = n & 3;
  const float* Wg = (n < 4) ? Wf : (n < 8) ? Wi : (n < 12) ? Wu : Wo;
  float wh0 = Wg[(EMB + 0) * NQ + q];
  float wh1 = Wg[(EMB + 1) * NQ + q];
  float wh2 = Wg[(EMB + 2) * NQ + q];
  float wh3 = Wg[(EMB + 3) * NQ + q];

  float hx0 = 0.f, hx1 = 0.f, hx2 = 0.f, hx3 = 0.f;
  float cx = 0.f;                       // cell state of unit q (replicated)
  bool is_u = (n >= 8) && (n < 12);

  for (int t = 0; t < SEQ; ++t) {
    float z = pre[t * 16 + n] + hx0 * wh0 + hx1 * wh1 + hx2 * wh2 + hx3 * wh3;
    float a = __cosf(z);                // RY(x) then RY(phi), measure Z
    float sg = 1.0f / (1.0f + __expf(-a));
    float th = tanhf(a);
    float act = is_u ? th : sg;

    float fq = __shfl(act,      q, 32);
    float iq = __shfl(act,  4 + q, 32);
    float uq = __shfl(act,  8 + q, 32);
    float oq = __shfl(act, 12 + q, 32);
    cx = fq * cx + iq * uq;
    float h = oq * tanhf(cx);

    hx0 = __shfl(h, 0, 32);
    hx1 = __shfl(h, 1, 32);
    hx2 = __shfl(h, 2, 32);
    hx3 = __shfl(h, 3, 32);
    if (lane < 4) hout[t * 4 + lane] = h;
  }
}

// ---------------------------------------------------------------------------
// K3: tag logits + log_softmax. One 64-thread block per timestep.
// ---------------------------------------------------------------------------
__global__ void __launch_bounds__(64) qlstm_tag(
    const float* __restrict__ hout, const float* __restrict__ Wt,
    const float* __restrict__ bt, float* __restrict__ out) {
  int t = blockIdx.x;
  int n = threadIdx.x;
  float h0 = hout[t * 4 + 0], h1 = hout[t * 4 + 1];
  float h2 = hout[t * 4 + 2], h3 = hout[t * 4 + 3];
  float l = bt[n] + h0 * Wt[n] + h1 * Wt[64 + n] + h2 * Wt[128 + n] + h3 * Wt[192 + n];

  __shared__ float red[64];
  red[n] = l;
  __syncthreads();
#pragma unroll
  for (int s = 32; s > 0; s >>= 1) {
    if (n < s) red[n] = fmaxf(red[n], red[n + s]);
    __syncthreads();
  }
  float mx = red[0];
  __syncthreads();
  float e = __expf(l - mx);
  red[n] = e;
  __syncthreads();
#pragma unroll
  for (int s = 32; s > 0; s >>= 1) {
    if (n < s) red[n] += red[n + s];
    __syncthreads();
  }
  float lse = mx + __logf(red[0]);
  out[(size_t)t * TAGS + n] = l - lse;
}

// ---------------------------------------------------------------------------
extern "C" void kernel_launch(void* const* d_in, const int* in_sizes, int n_in,
                              void* d_out, int out_size, void* d_ws, size_t ws_size,
                              hipStream_t stream) {
  const int*   sentence = (const int*)d_in[0];
  const float* emb      = (const float*)d_in[1];
  const float* Wf = (const float*)d_in[2],  *bf = (const float*)d_in[3],  *pf = (const float*)d_in[4];
  const float* Wi = (const float*)d_in[5],  *bi = (const float*)d_in[6],  *pi = (const float*)d_in[7];
  const float* Wu = (const float*)d_in[8],  *bu = (const float*)d_in[9],  *pu = (const float*)d_in[10];
  const float* Wo = (const float*)d_in[11], *bo = (const float*)d_in[12], *po = (const float*)d_in[13];
  const float* Wt = (const float*)d_in[14], *bt = (const float*)d_in[15];
  float* out = (float*)d_out;

  char* ws = (char*)d_ws;
  unsigned int* bfrag = (unsigned int*)ws;                    // 32*32*8*4   = 32 KB
  float*        pre   = (float*)(ws + (32u << 10));           // 2048*16*4   = 128 KB
  float*        hout  = (float*)(ws + (32u << 10) + (128u << 10)); // 2048*4*4 = 32 KB

  qlstm_pack_b  <<<KTILES, 32, 0, stream>>>(Wf, Wi, Wu, Wo, bfrag);
  qlstm_gemm_pre<<<MTILES, 32, 0, stream>>>(sentence, emb, bfrag,
                                            bf, bi, bu, bo, pf, pi, pu, po, pre);
  qlstm_scan    <<<1, 32, 0, stream>>>(pre, Wf, Wi, Wu, Wo, hout);
  qlstm_tag     <<<SEQ, 64, 0, stream>>>(hout, Wt, bt, out);
}
